// scAFGCC_76141180223878
// MI455X (gfx1250) — matrix operand order repaired and checked by reading, not compiled, not run on hardware
//
#include <hip/hip_runtime.h>
#include <hip/hip_bf16.h>

typedef __bf16 bf16;
typedef bf16  v16bf __attribute__((ext_vector_type(16)));
typedef bf16  v8bf  __attribute__((ext_vector_type(8)));
typedef float v8f   __attribute__((ext_vector_type(8)));
typedef int   v4i   __attribute__((vector_size(16)));
typedef v4i __attribute__((address_space(1)))* gv4i_p;   // global int4*
typedef v4i __attribute__((address_space(3)))* lv4i_p;   // LDS    int4*

#define DIM        256
#define TOPK       10
#define NCENT      10
#define NSEED      5
#define NITER      20
#define SELF_BOOST 10.0f
#define NMAX       8192

// ---------------------------------------------------------------------------
// Row L2-normalize: one wave (32 lanes) per row of 256 f32. Writes bf16 copy
// (for WMMA operands) and optionally an f32 copy (for k-means on teacher).
// ---------------------------------------------------------------------------
__global__ void __launch_bounds__(256)
k_normalize(const float* __restrict__ in, bf16* __restrict__ out_bf,
            float* __restrict__ out_f32, int n)
{
    const int wave = threadIdx.x >> 5;
    const int lane = threadIdx.x & 31;
    const int row  = blockIdx.x * 8 + wave;
    if (row >= n) return;

    const float* src = in + (size_t)row * DIM;
    float x[8];
    float ss = 0.0f;
#pragma unroll
    for (int i = 0; i < 8; ++i) { x[i] = src[lane * 8 + i]; ss += x[i] * x[i]; }
#pragma unroll
    for (int off = 16; off >= 1; off >>= 1) ss += __shfl_xor(ss, off);
    const float inv = rsqrtf(ss);

    v8bf o;
    float* fo = out_f32 ? out_f32 + (size_t)row * DIM : nullptr;
#pragma unroll
    for (int i = 0; i < 8; ++i) {
        float v = x[i] * inv;
        o[i] = (bf16)v;
        if (fo) fo[lane * 8 + i] = v;
    }
    *(v8bf*)(out_bf + (size_t)row * DIM + lane * 8) = o;
}

// ---------------------------------------------------------------------------
// sim = Sn @ Tn^T + SELF_BOOST*I   via v_wmma_f32_16x16x32_bf16.
// Block = 256 threads = 8 waves laid out 4(M) x 2(N); each wave owns a
// 32(M) x 64(N) tile: 2 A-frags x 4 B-frags -> 8 WMMAs / k-step with
// 2x fragment reuse in both directions (~21 FLOP per loaded byte).
// Block macro-tile = 128 x 128.
//   A frag (16x32, MxK): lane l<16 -> row M=l, K = kb..kb+15 (16 contiguous)
//                        lane l>=16 -> row M=l-16, kb offset +16.
//   B frag (32x16, KxN): lane l -> col N=l&15, 16 contiguous K at
//                        k + (l>=16 ? 16 : 0). B row = teacher row (S.T^T).
//   C frag: lane l -> N=l&15, VGPR v -> M = v + (l>=16 ? 8 : 0).
// ---------------------------------------------------------------------------
__global__ void __launch_bounds__(256)
k_gemm_sim(const bf16* __restrict__ A, const bf16* __restrict__ B,
           float* __restrict__ C, int n)
{
    const int lane  = threadIdx.x & 31;
    const int wave  = threadIdx.x >> 5;
    const int kh    = lane >> 4;        // 0/1 half-wave
    const int l15   = lane & 15;
    const int waveM = wave >> 1;        // 0..3
    const int waveN = wave & 1;         // 0..1

    const int mbase = blockIdx.y * 128 + waveM * 32;
    const int nbase = blockIdx.x * 128 + waveN * 64;

    const bf16* arow[2];
    arow[0] = A + (size_t)(mbase + l15) * DIM;
    arow[1] = A + (size_t)(mbase + 16 + l15) * DIM;
    const bf16* brow[4];
#pragma unroll
    for (int nt = 0; nt < 4; ++nt)
        brow[nt] = B + (size_t)(nbase + nt * 16 + l15) * DIM;

    v8f acc[2][4] = {};

    for (int k = 0; k < DIM; k += 32) {
        if (k + 32 < DIM) {               // gfx1250 global_prefetch_b8 (WGP scope)
            __builtin_prefetch(arow[0] + k + 32, 0, 3);
            __builtin_prefetch(arow[1] + k + 32, 0, 3);
            __builtin_prefetch(brow[0] + k + 32, 0, 3);
            __builtin_prefetch(brow[2] + k + 32, 0, 3);
        }
        const int akb = k + kh * 16;
        v16bf a[2];
#pragma unroll
        for (int mt = 0; mt < 2; ++mt) {
            v8bf alo = *(const v8bf*)(arow[mt] + akb);
            v8bf ahi = *(const v8bf*)(arow[mt] + akb + 8);
#pragma unroll
            for (int i = 0; i < 8; ++i) { a[mt][i] = alo[i]; a[mt][i + 8] = ahi[i]; }
        }
#pragma unroll
        for (int nt = 0; nt < 4; ++nt) {
            v16bf b = *(const v16bf*)(brow[nt] + k + kh * 16);
            acc[0][nt] = __builtin_amdgcn_wmma_f32_16x16x32_bf16(
                false, a[0], false, b, (short)0, acc[0][nt], false, false);
            acc[1][nt] = __builtin_amdgcn_wmma_f32_16x16x32_bf16(
                false, a[1], false, b, (short)0, acc[1][nt], false, false);
        }
    }

#pragma unroll
    for (int mt = 0; mt < 2; ++mt) {
#pragma unroll
        for (int nt = 0; nt < 4; ++nt) {
            const int col = nbase + nt * 16 + l15;
#pragma unroll
            for (int v = 0; v < 8; ++v) {
                const int row = mbase + mt * 16 + v + kh * 8;
                float val = acc[mt][nt][v];
                if (row == col) val += SELF_BOOST;
                C[(size_t)row * n + col] = val;
            }
        }
    }
}

// ---------------------------------------------------------------------------
// Per-row top-10: stage the 32KB row into LDS (gfx1250 async global->LDS,
// ASYNCcnt-tracked, no VGPR round-trip; falls back to plain staging), then
// 10 rounds of block argmax (value-desc, index-asc tiebreak a la lax.top_k).
// ---------------------------------------------------------------------------
__global__ void __launch_bounds__(256)
k_topk(const float* __restrict__ sim, int* __restrict__ knn_i,
       float* __restrict__ knn_f, int n)
{
    __shared__ float vals[NMAX];
    __shared__ float rv[256];
    __shared__ int   ri[256];

    const int row = blockIdx.x;
    const int t   = threadIdx.x;
    const float* src = sim + (size_t)row * n;

#if defined(__gfx1250__) && __has_builtin(__builtin_amdgcn_global_load_async_to_lds_b128)
    for (int j = t * 4; j < n; j += 256 * 4) {
        __builtin_amdgcn_global_load_async_to_lds_b128(
            (gv4i_p)(src + j), (lv4i_p)(&vals[j]), 0, 0);
    }
#if __has_builtin(__builtin_amdgcn_s_wait_asynccnt)
    __builtin_amdgcn_s_wait_asynccnt(0);
#else
    asm volatile("s_wait_asynccnt 0x0" ::: "memory");
#endif
#else
    for (int j = t; j < n; j += 256) vals[j] = src[j];
#endif
    __syncthreads();

    for (int sel = 0; sel < TOPK; ++sel) {
        float best = -__builtin_inff();
        int   bidx = 0x7fffffff;
        for (int j = t; j < n; j += 256) {
            float v = vals[j];
            if (v > best) { best = v; bidx = j; }
        }
        rv[t] = best; ri[t] = bidx;
        __syncthreads();
        for (int s = 128; s >= 1; s >>= 1) {
            if (t < s) {
                bool take = (rv[t + s] > rv[t]) ||
                            (rv[t + s] == rv[t] && ri[t + s] < ri[t]);
                if (take) { rv[t] = rv[t + s]; ri[t] = ri[t + s]; }
            }
            __syncthreads();
        }
        if (t == 0) {
            int w = ri[0];
            knn_i[row * TOPK + sel] = w;
            knn_f[row * TOPK + sel] = (float)w;
            vals[w] = -__builtin_inff();
        }
        __syncthreads();
    }
}

// ---------------------------------------------------------------------------
// k-means helpers
// ---------------------------------------------------------------------------
__global__ void k_zero_f(float* __restrict__ p, long long n)
{
    long long i = (long long)blockIdx.x * blockDim.x + threadIdx.x;
    if (i < n) p[i] = 0.0f;
}
__global__ void k_zero_u(unsigned* __restrict__ p, long long n)
{
    long long i = (long long)blockIdx.x * blockDim.x + threadIdx.x;
    if (i < n) p[i] = 0u;
}

// centroid init: hash-picked rows (stand-in for jax.random.choice)
__global__ void k_kminit(const float* __restrict__ x, float* __restrict__ cent,
                         int seed, int n)
{
    const int c = blockIdx.x;
    unsigned h = (unsigned)(seed + 1) * 2654435761u ^ ((unsigned)c * 40503u + 12345u);
    h ^= h >> 13; h *= 2246822519u; h ^= h >> 16;
    const int idx = (int)(h % (unsigned)n);
    for (int d = threadIdx.x; d < DIM; d += blockDim.x)
        cent[c * DIM + d] = x[(size_t)idx * DIM + d];
}

// assign: one wave per point; argmin_c (|c|^2 - 2 x.c)
__global__ void __launch_bounds__(256)
k_assign(const float* __restrict__ x, const float* __restrict__ cent,
         int* __restrict__ labels, int n)
{
    __shared__ float sc[NCENT * DIM];
    __shared__ float scn[NCENT];
    for (int i = threadIdx.x; i < NCENT * DIM; i += 256) sc[i] = cent[i];
    __syncthreads();
    if (threadIdx.x < NCENT) {
        float s = 0.0f;
        for (int d = 0; d < DIM; ++d) { float v = sc[threadIdx.x * DIM + d]; s += v * v; }
        scn[threadIdx.x] = s;
    }
    __syncthreads();

    const int wave = threadIdx.x >> 5, lane = threadIdx.x & 31;
    const int p = blockIdx.x * 8 + wave;
    if (p >= n) return;

    float xv[8];
    const float* xp = x + (size_t)p * DIM;
#pragma unroll
    for (int i = 0; i < 8; ++i) xv[i] = xp[lane * 8 + i];

    float bestd = __builtin_inff();
    int   bestc = 0;
    for (int c = 0; c < NCENT; ++c) {
        float dot = 0.0f;
#pragma unroll
        for (int i = 0; i < 8; ++i) dot += xv[i] * sc[c * DIM + lane * 8 + i];
#pragma unroll
        for (int off = 16; off >= 1; off >>= 1) dot += __shfl_xor(dot, off);
        float d2 = scn[c] - 2.0f * dot;   // +|x|^2 constant -> argmin unchanged
        if (d2 < bestd) { bestd = d2; bestc = c; }
    }
    if (lane == 0) labels[p] = bestc;
}

// accumulate: thread per point; LDS-staged sums then global atomics
__global__ void __launch_bounds__(256)
k_accum(const float* __restrict__ x, const int* __restrict__ labels,
        float* __restrict__ sums, float* __restrict__ cnts, int n)
{
    __shared__ float ls[NCENT * DIM];
    __shared__ float lc[NCENT];
    for (int i = threadIdx.x; i < NCENT * DIM; i += 256) ls[i] = 0.0f;
    if (threadIdx.x < NCENT) lc[threadIdx.x] = 0.0f;
    __syncthreads();

    const int p = blockIdx.x * 256 + threadIdx.x;
    if (p < n) {
        const int lab = labels[p];
        const float* xp = x + (size_t)p * DIM;
        for (int d = 0; d < DIM; ++d) atomicAdd(&ls[lab * DIM + d], xp[d]);
        atomicAdd(&lc[lab], 1.0f);
    }
    __syncthreads();
    for (int i = threadIdx.x; i < NCENT * DIM; i += 256)
        if (ls[i] != 0.0f) atomicAdd(&sums[i], ls[i]);
    if (threadIdx.x < NCENT && lc[threadIdx.x] != 0.0f)
        atomicAdd(&cnts[threadIdx.x], lc[threadIdx.x]);
}

// update: grid NCENT x block DIM
__global__ void k_update(float* __restrict__ cent, const float* __restrict__ sums,
                         const float* __restrict__ cnts)
{
    const int c = blockIdx.x, d = threadIdx.x;
    const float cnt = cnts[c];
    if (cnt > 0.0f) cent[c * DIM + d] = sums[c * DIM + d] / fmaxf(cnt, 1.0f);
}

// ---------------------------------------------------------------------------
// adjacency bitmask + final pos_mask
// ---------------------------------------------------------------------------
__global__ void k_edges(const int* __restrict__ ei, unsigned* __restrict__ adj,
                        int e, int n)
{
    const int i = blockIdx.x * 256 + threadIdx.x;
    if (i >= e) return;
    const int s = ei[i];
    const int d = ei[(size_t)e + i];
    atomicOr(&adj[(size_t)s * (n >> 5) + (d >> 5)], 1u << (d & 31));
}

__global__ void k_posmask(const int* __restrict__ knn, const int* __restrict__ labels,
                          const unsigned* __restrict__ adj, float* __restrict__ pos,
                          int n)
{
    const int idx = blockIdx.x * 256 + threadIdx.x;
    if (idx >= n * TOPK) return;
    const int i = idx / TOPK;
    const int j = knn[idx];
    const bool in_adj = (adj[(size_t)i * (n >> 5) + (j >> 5)] >> (j & 31)) & 1u;
    bool close = false;
#pragma unroll
    for (int s = 0; s < NSEED; ++s)
        close = close || (labels[s * n + i] == labels[s * n + j]);
    pos[idx] = (in_adj || close) ? 1.0f : 0.0f;
}

// ---------------------------------------------------------------------------
extern "C" void kernel_launch(void* const* d_in, const int* in_sizes, int n_in,
                              void* d_out, int out_size, void* d_ws, size_t ws_size,
                              hipStream_t stream)
{
    const float* student = (const float*)d_in[0];
    const float* teacher = (const float*)d_in[1];
    const int*   edges   = (const int*)d_in[2];     // [2, e] flat
    const int n = in_sizes[0] / DIM;                // 8192
    const int e = in_sizes[2] / 2;                  // 131072
    (void)n_in; (void)out_size; (void)ws_size;

    float* sim     = (float*)d_out;
    float* out_knn = sim + (size_t)n * n;
    float* out_pos = out_knn + (size_t)n * TOPK;

    char* ws = (char*)d_ws;
    bf16*     sbf    = (bf16*)ws;     ws += (size_t)n * DIM * 2;
    bf16*     tbf    = (bf16*)ws;     ws += (size_t)n * DIM * 2;
    float*    tf     = (float*)ws;    ws += (size_t)n * DIM * 4;
    float*    cent   = (float*)ws;    ws += (size_t)NCENT * DIM * 4;
    float*    sums   = (float*)ws;    ws += (size_t)NCENT * DIM * 4;
    float*    cnts   = (float*)ws;    ws += 256;
    int*      labels = (int*)ws;      ws += (size_t)NSEED * n * 4;
    int*      knn_i  = (int*)ws;      ws += (size_t)n * TOPK * 4;
    unsigned* adj    = (unsigned*)ws; ws += (size_t)n * (n >> 5) * 4;

    // 1) normalize (bf16 WMMA operands; f32 teacher copy for k-means)
    k_normalize<<<n / 8, 256, 0, stream>>>(student, sbf, nullptr, n);
    k_normalize<<<n / 8, 256, 0, stream>>>(teacher, tbf, tf, n);

    // 2) sim = Sn @ Tn^T + 10I  (bf16 WMMA, f32 accumulate, 128x128 blocks)
    dim3 grid(n / 128, n / 128);
    k_gemm_sim<<<grid, 256, 0, stream>>>(sbf, tbf, sim, n);

    // 3) per-row top-10
    k_topk<<<n, 256, 0, stream>>>(sim, knn_i, out_knn, n);

    // 4) k-means consensus labels (5 seeds x 20 iters)
    const long long nsums = (long long)NCENT * DIM;
    for (int s = 0; s < NSEED; ++s) {
        k_kminit<<<NCENT, 256, 0, stream>>>(tf, cent, s, n);
        for (int it = 0; it < NITER; ++it) {
            k_zero_f<<<(int)((nsums + 255) / 256), 256, 0, stream>>>(sums, nsums);
            k_zero_f<<<1, 256, 0, stream>>>(cnts, NCENT);
            k_assign<<<n / 8, 256, 0, stream>>>(tf, cent, labels + (size_t)s * n, n);
            k_accum<<<(n + 255) / 256, 256, 0, stream>>>(tf, labels + (size_t)s * n, sums, cnts, n);
            k_update<<<NCENT, DIM, 0, stream>>>(cent, sums, cnts);
        }
        k_assign<<<n / 8, 256, 0, stream>>>(tf, cent, labels + (size_t)s * n, n);
    }

    // 5) adjacency bits + fused pos_mask
    const long long nwords = (long long)n * (n >> 5);
    k_zero_u<<<(int)((nwords + 255) / 256), 256, 0, stream>>>(adj, nwords);
    k_edges<<<(e + 255) / 256, 256, 0, stream>>>(edges, adj, e, n);
    k_posmask<<<(n * TOPK + 255) / 256, 256, 0, stream>>>(knn_i, labels, adj, out_pos, n);
}
